// SABLock_76278619177194
// MI455X (gfx1250) — matrix operand order repaired and checked by reading, not compiled, not run on hardware
//
#include <hip/hip_runtime.h>

// ---------------------------------------------------------------------------
// MI455X (gfx1250) self-attention block.
//   fx = Wf x + bf ; gx = Wg x + bg ; hx = Wh x + bh        (bf16 in ws)
//   S  = Fx^T Gx ; A = softmax_rows(S) ; O = Hx A           (per b,c 64x64)
//   y  = Wv O + bv                                          (f32 out)
// All matmuls use v_wmma_f32_16x16x32_bf16 (wave32, 16x16 tiles).
// Weights pre-converted to bf16 once (512 KB, L2-resident) so the GEMM
// inner loops do no f32->bf16 VALU work.
// Workspace: fx (reused for O) + gx + hx bf16 (96 MiB) + weights bf16 (512 KB).
// ---------------------------------------------------------------------------

typedef __attribute__((ext_vector_type(16))) __bf16 v16bf;
typedef __attribute__((ext_vector_type(8)))  float  v8f;

union FragBF {                 // one WMMA bf16 operand fragment (per lane)
  v16bf          v;
  unsigned short us[16];
  uint4          q[2];
};
union F4U { float4 q; float f[4]; };
union U2U { uint2  q; unsigned short s[4]; };

__device__ __forceinline__ unsigned short f2us(float f) {  // f32 -> bf16 (RNE)
  union { float f; unsigned u; } v; v.f = f;
  unsigned r = v.u + 0x7FFFu + ((v.u >> 16) & 1u);
  return (unsigned short)(r >> 16);
}

#define WMMA_BF16(A, B, C) \
  __builtin_amdgcn_wmma_f32_16x16x32_bf16(false, (A), false, (B), (short)0, (C), false, false)

// ---------------------------------------------------------------------------
// Kernel 0: one-shot weight conversion f32 -> bf16 (row-major preserved).
// dst holds 4 slabs of 256x256: [0]=Wf [1]=Wg [2]=Wh [3]=Wv.
// ---------------------------------------------------------------------------
__global__ __launch_bounds__(256) void cvt_w_kernel(
    const float* __restrict__ wf, const float* __restrict__ wg,
    const float* __restrict__ wh, const float* __restrict__ wv,
    unsigned short* __restrict__ dst)
{
  const float* W[4] = {wf, wg, wh, wv};
  const float* src = W[blockIdx.y];
  unsigned short* d = dst + (size_t)blockIdx.y * 65536;
  int i = (blockIdx.x * 256 + threadIdx.x) * 4;
  F4U a; a.q = *(const float4*)(src + i);
  uint2 p;
  p.x = (unsigned)f2us(a.f[0]) | ((unsigned)f2us(a.f[1]) << 16);
  p.y = (unsigned)f2us(a.f[2]) | ((unsigned)f2us(a.f[3]) << 16);
  *(uint2*)(d + i) = p;
}

// ---------------------------------------------------------------------------
// A-fragment loader: bf16 row-major W, tile (row0..row0+15, kk..kk+31).
// Per documented 16-bit A layout: lane half h -> K runs [kk+8h, +7] and
// [kk+16+8h, +7]; both contiguous 16B -> two b128 loads, no conversion.
// ---------------------------------------------------------------------------
__device__ __forceinline__ void load_a_frag(FragBF& A, const unsigned short* Wb,
                                            int row0, int kk, int l16, int lhalf)
{
  const unsigned short* wr = Wb + (size_t)(row0 + l16) * 256 + kk + lhalf * 8;
  A.q[0] = *(const uint4*)(wr);
  A.q[1] = *(const uint4*)(wr + 16);
}

// ---------------------------------------------------------------------------
// Kernel 1: fused 3x conv1x1 (f,g,h). Grid (64 n-tiles, 16 batches), 256 thr.
// Per block: load X tile [K=256][N=64] once as bf16 into LDS [N][K], then
// 3 GEMM passes over Wf/Wg/Wh. 8 waves, each owns 32 out-rows x 64 cols.
// ---------------------------------------------------------------------------
__global__ __launch_bounds__(256) void conv_in_kernel(
    const float* __restrict__ x,
    const unsigned short* __restrict__ wb,          // bf16 weights (4 slabs)
    const float* __restrict__ bfv, const float* __restrict__ bgv,
    const float* __restrict__ bhv,
    unsigned short* __restrict__ fxb, unsigned short* __restrict__ gxb,
    unsigned short* __restrict__ hxb)
{
  __shared__ unsigned short Bs[64][264];   // [N][K], pad -> 528B row stride

  const int t  = threadIdx.x;
  const int b  = blockIdx.y;
  const int n0 = blockIdx.x * 64;
  const float* xb = x + (size_t)b * 256 * 4096 + n0;

  // ---- load + transpose X tile (f32 -> bf16), 4x4 micro-blocks ----
  #pragma unroll
  for (int blk = 0; blk < 4; ++blk) {
    int bid = t + blk * 256;               // 0..1023
    int rg  = bid & 63, cg = bid >> 6;     // 64 K-groups x 16 N-groups
    int i0 = rg * 4,  c0 = cg * 4;
    F4U r[4];
    #pragma unroll
    for (int j = 0; j < 4; ++j)
      r[j].q = *(const float4*)(xb + (size_t)(i0 + j) * 4096 + c0);
    #pragma unroll
    for (int c = 0; c < 4; ++c) {
      uint2 p;
      p.x = (unsigned)f2us(r[0].f[c]) | ((unsigned)f2us(r[1].f[c]) << 16);
      p.y = (unsigned)f2us(r[2].f[c]) | ((unsigned)f2us(r[3].f[c]) << 16);
      *(uint2*)&Bs[c0 + c][i0] = p;
    }
  }
  __syncthreads();

  const int wave  = t >> 5, lane = t & 31;
  const int lhalf = lane >> 4, l16 = lane & 15;
  const int mbase = wave * 32;

  const float* Bias[3] = {bfv, bgv, bhv};
  unsigned short* Dst[3] = {fxb, gxb, hxb};

  #pragma unroll
  for (int which = 0; which < 3; ++which) {
    const unsigned short* Wb = wb + (size_t)which * 65536;
    v8f acc[2][4] = {};

    for (int kk = 0; kk < 256; kk += 32) {
      FragBF A[2];
      load_a_frag(A[0], Wb, mbase,      kk, l16, lhalf);
      load_a_frag(A[1], Wb, mbase + 16, kk, l16, lhalf);
      FragBF Bf[4];
      #pragma unroll
      for (int nj = 0; nj < 4; ++nj) {
        const unsigned short* bp = &Bs[nj * 16 + l16][kk + lhalf * 8];
        Bf[nj].q[0] = *(const uint4*)(bp);
        Bf[nj].q[1] = *(const uint4*)(bp + 16);
      }
      #pragma unroll
      for (int mi = 0; mi < 2; ++mi)
        #pragma unroll
        for (int nj = 0; nj < 4; ++nj)
          acc[mi][nj] = WMMA_BF16(A[mi].v, Bf[nj].v, acc[mi][nj]);
    }

    // bias + store bf16
    float bvals[2][8];
    #pragma unroll
    for (int mi = 0; mi < 2; ++mi) {
      const float* bp = Bias[which] + mbase + mi * 16 + lhalf * 8;
      F4U b0, b1; b0.q = *(const float4*)bp; b1.q = *(const float4*)(bp + 4);
      #pragma unroll
      for (int e = 0; e < 4; ++e) { bvals[mi][e] = b0.f[e]; bvals[mi][4 + e] = b1.f[e]; }
    }
    unsigned short* db = Dst[which] + (size_t)b * 256 * 4096 + n0;
    #pragma unroll
    for (int mi = 0; mi < 2; ++mi)
      #pragma unroll
      for (int nj = 0; nj < 4; ++nj)
        #pragma unroll
        for (int r = 0; r < 8; ++r) {
          int row = mbase + mi * 16 + lhalf * 8 + r;
          db[(size_t)row * 4096 + nj * 16 + l16] = f2us(acc[mi][nj][r] + bvals[mi][r]);
        }
  }
}

// ---------------------------------------------------------------------------
// Kernel 2: attention per (b,c). Block = 256 thr, one 64x64 problem in LDS.
// S = Fx^T Gx (WMMA) -> f32 LDS -> row softmax -> Att^T bf16 -> O = Hx Att.
// O overwrites the fx slab (read fully into LDS before any write).
// ---------------------------------------------------------------------------
__global__ __launch_bounds__(256) void attn_kernel(
    const unsigned short* __restrict__ gxb,
    const unsigned short* __restrict__ hxb,
    unsigned short* fxb)
{
  __shared__ unsigned short FxT[64][72];   // [w][h]  (A of S)
  __shared__ unsigned short GsT[64][72];   // [v][h]  (B of S, [N][K])
  __shared__ unsigned short Hs [64][72];   // [h][w]  (A of O)
  __shared__ unsigned short AtT[64][72];   // [v][w]  (B of O, [N][K])
  __shared__ float          Sf [64][68];   // scores f32

  const int t  = threadIdx.x;
  const int bc = blockIdx.x;                      // b*256 + c
  const unsigned short* fslab = fxb + (size_t)bc * 4096;
  const unsigned short* gslab = gxb + (size_t)bc * 4096;
  const unsigned short* hslab = hxb + (size_t)bc * 4096;

  // ---- load: Fx,Gx transposed (4x4 blocks), Hx direct ----
  {
    int rg = t >> 4, cg = t & 15;
    int h0 = rg * 4, w0 = cg * 4;
    U2U fr[4], gr[4];
    #pragma unroll
    for (int j = 0; j < 4; ++j) {
      fr[j].q = *(const uint2*)(fslab + (h0 + j) * 64 + w0);
      gr[j].q = *(const uint2*)(gslab + (h0 + j) * 64 + w0);
    }
    #pragma unroll
    for (int c = 0; c < 4; ++c) {
      uint2 pf; pf.x = (unsigned)fr[0].s[c] | ((unsigned)fr[1].s[c] << 16);
                pf.y = (unsigned)fr[2].s[c] | ((unsigned)fr[3].s[c] << 16);
      uint2 pg; pg.x = (unsigned)gr[0].s[c] | ((unsigned)gr[1].s[c] << 16);
                pg.y = (unsigned)gr[2].s[c] | ((unsigned)gr[3].s[c] << 16);
      *(uint2*)&FxT[w0 + c][h0] = pf;
      *(uint2*)&GsT[w0 + c][h0] = pg;
    }
    int hr = t >> 2, qd = t & 3;
    *(uint4*)&Hs[hr][qd * 16]     = *(const uint4*)(hslab + hr * 64 + qd * 16);
    *(uint4*)&Hs[hr][qd * 16 + 8] = *(const uint4*)(hslab + hr * 64 + qd * 16 + 8);
  }
  __syncthreads();

  const int wave = t >> 5, lane = t & 31;
  const int lhalf = lane >> 4, l16 = lane & 15;

  // ---- S = Fx^T Gx : 16 tiles, 2 per wave ----
  #pragma unroll
  for (int j = 0; j < 2; ++j) {
    int tid = wave * 2 + j;
    int m0 = (tid >> 2) * 16, nv = (tid & 3) * 16;
    v8f acc = {};
    #pragma unroll
    for (int kk = 0; kk < 64; kk += 32) {
      FragBF A, B;
      const unsigned short* ap = &FxT[m0 + l16][kk + lhalf * 8];
      A.q[0] = *(const uint4*)(ap);
      A.q[1] = *(const uint4*)(ap + 16);
      const unsigned short* bp = &GsT[nv + l16][kk + lhalf * 8];
      B.q[0] = *(const uint4*)(bp);
      B.q[1] = *(const uint4*)(bp + 16);
      acc = WMMA_BF16(A.v, B.v, acc);
    }
    #pragma unroll
    for (int r = 0; r < 8; ++r)
      Sf[m0 + lhalf * 8 + r][nv + l16] = acc[r];
  }
  __syncthreads();

  // ---- row softmax (64 rows, threads 0..63), write Att^T bf16 ----
  if (t < 64) {
    float mx = -3.4e38f;
    for (int v = 0; v < 64; ++v) mx = fmaxf(mx, Sf[t][v]);
    float s = 0.f;
    for (int v = 0; v < 64; ++v) { float e = __expf(Sf[t][v] - mx); Sf[t][v] = e; s += e; }
    float inv = 1.0f / s;
    for (int v = 0; v < 64; ++v) AtT[v][t] = f2us(Sf[t][v] * inv);
  }
  __syncthreads();

  // ---- O = Hx * Att, overwrite fx slab ----
  unsigned short* oslab = fxb + (size_t)bc * 4096;
  #pragma unroll
  for (int j = 0; j < 2; ++j) {
    int tid = wave * 2 + j;
    int m0 = (tid >> 2) * 16, nv = (tid & 3) * 16;
    v8f acc = {};
    #pragma unroll
    for (int kk = 0; kk < 64; kk += 32) {
      FragBF A, B;
      const unsigned short* ap = &Hs[m0 + l16][kk + lhalf * 8];
      A.q[0] = *(const uint4*)(ap);
      A.q[1] = *(const uint4*)(ap + 16);
      const unsigned short* bp = &AtT[nv + l16][kk + lhalf * 8];
      B.q[0] = *(const uint4*)(bp);
      B.q[1] = *(const uint4*)(bp + 16);
      acc = WMMA_BF16(A.v, B.v, acc);
    }
    #pragma unroll
    for (int r = 0; r < 8; ++r)
      oslab[(m0 + lhalf * 8 + r) * 64 + nv + l16] = f2us(acc[r]);
  }
}

// ---------------------------------------------------------------------------
// Kernel 3: y = Wv * O + bv (bf16 in, f32 out). Grid (64, 16), 256 thr.
// ---------------------------------------------------------------------------
__global__ __launch_bounds__(256) void conv_out_kernel(
    const unsigned short* __restrict__ obuf,
    const unsigned short* __restrict__ wvb,         // bf16 Wv
    const float* __restrict__ bvv,
    float* __restrict__ y)
{
  __shared__ unsigned short Bs[64][264];

  const int t  = threadIdx.x;
  const int b  = blockIdx.y;
  const int n0 = blockIdx.x * 64;
  const unsigned short* ob = obuf + (size_t)b * 256 * 4096 + n0;

  #pragma unroll
  for (int blk = 0; blk < 4; ++blk) {
    int bid = t + blk * 256;
    int rg = bid & 63, cg = bid >> 6;
    int i0 = rg * 4, c0 = cg * 4;
    U2U r[4];
    #pragma unroll
    for (int j = 0; j < 4; ++j)
      r[j].q = *(const uint2*)(ob + (size_t)(i0 + j) * 4096 + c0);
    #pragma unroll
    for (int c = 0; c < 4; ++c) {
      uint2 p; p.x = (unsigned)r[0].s[c] | ((unsigned)r[1].s[c] << 16);
               p.y = (unsigned)r[2].s[c] | ((unsigned)r[3].s[c] << 16);
      *(uint2*)&Bs[c0 + c][i0] = p;
    }
  }
  __syncthreads();

  const int wave = t >> 5, lane = t & 31;
  const int lhalf = lane >> 4, l16 = lane & 15;
  const int mbase = wave * 32;
  v8f acc[2][4] = {};

  for (int kk = 0; kk < 256; kk += 32) {
    FragBF A[2];
    load_a_frag(A[0], wvb, mbase,      kk, l16, lhalf);
    load_a_frag(A[1], wvb, mbase + 16, kk, l16, lhalf);
    FragBF Bf[4];
    #pragma unroll
    for (int nj = 0; nj < 4; ++nj) {
      const unsigned short* bp = &Bs[nj * 16 + l16][kk + lhalf * 8];
      Bf[nj].q[0] = *(const uint4*)(bp);
      Bf[nj].q[1] = *(const uint4*)(bp + 16);
    }
    #pragma unroll
    for (int mi = 0; mi < 2; ++mi)
      #pragma unroll
      for (int nj = 0; nj < 4; ++nj)
        acc[mi][nj] = WMMA_BF16(A[mi].v, Bf[nj].v, acc[mi][nj]);
  }

  float bvals[2][8];
  #pragma unroll
  for (int mi = 0; mi < 2; ++mi) {
    const float* bp = bvv + mbase + mi * 16 + lhalf * 8;
    F4U b0, b1; b0.q = *(const float4*)bp; b1.q = *(const float4*)(bp + 4);
    #pragma unroll
    for (int e = 0; e < 4; ++e) { bvals[mi][e] = b0.f[e]; bvals[mi][4 + e] = b1.f[e]; }
  }
  float* yb = y + (size_t)b * 256 * 4096 + n0;
  #pragma unroll
  for (int mi = 0; mi < 2; ++mi)
    #pragma unroll
    for (int nj = 0; nj < 4; ++nj)
      #pragma unroll
      for (int r = 0; r < 8; ++r) {
        int row = mbase + mi * 16 + lhalf * 8 + r;
        yb[(size_t)row * 4096 + nj * 16 + l16] = acc[mi][nj][r] + bvals[mi][r];
      }
}

// ---------------------------------------------------------------------------
extern "C" void kernel_launch(void* const* d_in, const int* in_sizes, int n_in,
                              void* d_out, int out_size, void* d_ws, size_t ws_size,
                              hipStream_t stream) {
  const float* x  = (const float*)d_in[0];
  const float* wf = (const float*)d_in[1];
  const float* bf = (const float*)d_in[2];
  const float* wg = (const float*)d_in[3];
  const float* bg = (const float*)d_in[4];
  const float* wh = (const float*)d_in[5];
  const float* bh = (const float*)d_in[6];
  const float* wv = (const float*)d_in[7];
  const float* bv = (const float*)d_in[8];
  float* out = (float*)d_out;

  const size_t SLAB = (size_t)16 * 256 * 4096;      // 16,777,216 bf16 elems
  unsigned short* fxb = (unsigned short*)d_ws;      // fx, later O (aliased)
  unsigned short* gxb = fxb + SLAB;
  unsigned short* hxb = gxb + SLAB;
  unsigned short* wbb = hxb + SLAB;                 // 4 x 65536 bf16 weights

  cvt_w_kernel   <<<dim3(64, 4),  256, 0, stream>>>(wf, wg, wh, wv, wbb);
  conv_in_kernel <<<dim3(64, 16), 256, 0, stream>>>(x, wbb, bf, bg, bh,
                                                    fxb, gxb, hxb);
  attn_kernel    <<<dim3(4096),   256, 0, stream>>>(gxb, hxb, fxb);
  conv_out_kernel<<<dim3(64, 16), 256, 0, stream>>>(fxb, wbb + 3 * 65536, bv, out);
}
